// MFN_10479720202421
// MI455X (gfx1250) — matrix operand order repaired and checked by reading, not compiled, not run on hardware
//
// MFN (Memory Fusion Network) fused persistent kernel for MI455X (gfx1250).
//
// Strategy (compute-bound, WMMA-centric):
//  * 2048 batch rows / 16 = 128 persistent workgroups; each WG owns one 16-row
//    M-tile for all 128 timesteps. No inter-WG traffic; all recurrent state in LDS.
//  * Every GEMM (y = act @ W^T) runs on v_wmma_f32_16x16x32_f16 (f16 in, f32 acc).
//  * Weights are pre-packed ONCE (pack_weights kernel) into the exact per-lane
//    B-fragment layout of the ISA, so the inner loop is 2x global_load_b128 +
//    v_wmma per K-tile, streamed from L2 (weights ~1.2MB packed, fully L2-resident).
//  * KT and NT are template params: each wave's N-tiles become compile-time
//    immediate offsets (no EXEC masking / pointer selects around WMMA), and all
//    tiles-per-wave accumulate concurrently (up to 8 independent WMMA chains).
//  * A-fragments built from LDS with float4 loads (2 contiguous 8-float runs per
//    lane) and held in VGPRs for all K-tiles.
//  * LDS ~179KB/WG (<320KB WGP limit) -> 1 WG per WGP, 1 wave per SIMD, full
//    VGPR budget per wave.
#include <hip/hip_runtime.h>
#include <cstdint>

typedef __attribute__((ext_vector_type(16))) _Float16 v16h;
typedef __attribute__((ext_vector_type(8)))  float    v8f;

#define DEVINL __device__ __forceinline__

// ---------------- dims ----------------
#define T_STEPS 128
#define NBATCH  2048
#define D_L 300
#define D_A 74
#define D_V 36
#define D_X 410          // D_L + D_A + D_V
#define DH_L 128
#define DH_AV 32
#define TOTAL_H 192      // 128+32+32
#define MEMD 128
#define GAMMA_IN 320     // 192+128
// padded K (multiples of 32)
#define KL_PAD 448       // 300 + 128 -> 428 -> 448 (KT=14)
#define KA_PAD 128       // 74 + 32 -> 106 -> 128   (KT=4)
#define KV_PAD 96        // 36 + 32 -> 68  -> 96    (KT=3)

DEVINL float sigf(float x) { return 1.0f / (1.0f + expf(-x)); }

DEVINL v16h load_bfrag(const uint32_t* __restrict__ p) {
  union { uint4 u[2]; v16h h; } cv;
  const uint4* q = (const uint4*)p;
  cv.u[0] = q[0];
  cv.u[1] = q[1];
  return cv.h;
}

DEVINL void store_ctile(float* __restrict__ dst, int stride,
                        const float* __restrict__ bias,
                        const float* __restrict__ bias2,
                        int nt, v8f c, bool relu, int col, int mbase) {
  int n = nt * 16 + col;
  float b = bias ? bias[n] : 0.0f;
  if (bias2) b += bias2[n];
#pragma unroll
  for (int r = 0; r < 8; ++r) {
    float v = c[r] + b;
    if (relu) v = fmaxf(v, 0.0f);
    // C layout: VGPR r, lanes 0-15 -> M=r; lanes 16-31 -> M=r+8
    dst[(mbase + r) * stride + n] = v;
  }
}

// One workgroup GEMM: dst[16][NT*16] = acts[16][KT*32] @ W^T (+bias, opt relu).
// acts: f32 in LDS, row stride actStride (multiple of 8 floats), zero-padded to
// KT*32 columns. wfrag: packed f16 B fragments, frag (nt*KT + kt), 256 dwords.
// 4 waves split N-tiles statically (nt = wave + 4*i); all tiles of a wave
// accumulate concurrently -> TPW independent WMMA dependency chains.
template <int KT, int NT>
DEVINL void wg_gemm(const uint32_t* __restrict__ wfrag,
                    const float* __restrict__ acts, int actStride,
                    float* __restrict__ dst, int dstStride,
                    const float* __restrict__ bias,
                    const float* __restrict__ bias2,
                    bool relu, int lane, int wave) {
  static_assert(NT % 4 == 0, "NT must be a multiple of 4");
  constexpr int TPW = NT / 4;  // tiles per wave (1..8)
  const int m  = lane & 15;
  const int hs = lane >> 4;
  // Build A fragments for all K tiles; held in VGPRs.
  // A layout (16-bit A 16x32): lane row m; lanes<16 hold K {0..7,16..23},
  // lanes>=16 hold K {8..15,24..31}; dword j: k = (j>=4?16:0)+hs*8+(j&3)*2.
  // Per lane that is runs ar[0..7] and ar[16..23]: 4x float4 LDS loads.
  v16h a[KT];
#pragma unroll
  for (int kt = 0; kt < KT; ++kt) {
    const float* ar = acts + m * actStride + kt * 32 + hs * 8;  // 32B aligned
    const float4 f0 = *(const float4*)(ar);
    const float4 f1 = *(const float4*)(ar + 4);
    const float4 f2 = *(const float4*)(ar + 16);
    const float4 f3 = *(const float4*)(ar + 20);
    a[kt][0]  = (_Float16)f0.x; a[kt][1]  = (_Float16)f0.y;
    a[kt][2]  = (_Float16)f0.z; a[kt][3]  = (_Float16)f0.w;
    a[kt][4]  = (_Float16)f1.x; a[kt][5]  = (_Float16)f1.y;
    a[kt][6]  = (_Float16)f1.z; a[kt][7]  = (_Float16)f1.w;
    a[kt][8]  = (_Float16)f2.x; a[kt][9]  = (_Float16)f2.y;
    a[kt][10] = (_Float16)f2.z; a[kt][11] = (_Float16)f2.w;
    a[kt][12] = (_Float16)f3.x; a[kt][13] = (_Float16)f3.y;
    a[kt][14] = (_Float16)f3.z; a[kt][15] = (_Float16)f3.w;
  }
  // Wave base: tile nt = wave + 4*i at byte offset i*(4*KT*1024) from base.
  const uint32_t* base = wfrag + (size_t)wave * KT * 256 + lane * 8;
  v8f acc[TPW];
#pragma unroll
  for (int i = 0; i < TPW; ++i) { v8f z = {}; acc[i] = z; }
#pragma unroll
  for (int kt = 0; kt < KT; ++kt) {
#pragma unroll
    for (int i = 0; i < TPW; ++i) {
      v16h b = load_bfrag(base + ((size_t)i * 4 * KT + kt) * 256);
      acc[i] = __builtin_amdgcn_wmma_f32_16x16x32_f16(false, a[kt], false, b,
                                                      (short)0, acc[i], false, false);
    }
  }
  const int mbase = hs * 8;
#pragma unroll
  for (int i = 0; i < TPW; ++i)
    store_ctile(dst, dstStride, bias, bias2, wave + 4 * i, acc[i], relu, m, mbase);
}

// ------------- weight pre-pack: f32 row-major W[O][K0(+K1 concat)] ->
// f16 B fragments in WMMA per-lane layout. One 32-thread block per fragment.
// B layout (16-bit B 32x16): lanes 0-15 hold K {0..7,16..23} of column `lane`,
// lanes 16-31 hold K {8..15,24..31}; dword j: k = (j>=4?16:0)+hs*8+(j&3)*2 (+0,+1).
__global__ void pack_weights(const float* __restrict__ src0, int K0,
                             const float* __restrict__ src1, int K1,
                             int O, int KT, uint32_t* __restrict__ dst) {
  const int tile = blockIdx.x;       // nt*KT + kt
  const int kt = tile % KT;
  const int nt = tile / KT;
  const int lane = threadIdx.x;
  const int m  = lane & 15;
  const int hs = lane >> 4;
  const int col = nt * 16 + m;
  uint32_t* out = dst + (size_t)tile * 256 + lane * 8;
#pragma unroll
  for (int j = 0; j < 8; ++j) {
    uint32_t w = 0;
#pragma unroll
    for (int e = 0; e < 2; ++e) {
      const int k = kt * 32 + (j >> 2) * 16 + hs * 8 + ((2 * j + e) & 7);
      float v = 0.0f;
      if (col < O) {
        if (k < K0)            v = src0[(size_t)col * K0 + k];
        else if (k < K0 + K1)  v = src1[(size_t)col * K1 + (k - K0)];
      }
      union { _Float16 h; uint16_t u; } cv;
      cv.h = (_Float16)v;
      w |= (uint32_t)cv.u << (16 * e);
    }
    out[j] = w;
  }
}

struct MfnParams {
  const float* x;
  const float *bihL, *bhhL, *bihA, *bhhA, *bihV, *bhhV;
  const float *attB1, *attB2, *nnpB1, *nnpB2;
  const float *g1B1, *g1B2, *g2B1, *g2B2;
  const float *outB1, *outW2, *outB2;
  const uint32_t *WL, *WA, *WV, *ATT1, *ATT2, *NNP1, *NNP2;
  const uint32_t *G1A, *G1B, *G2A, *G2B, *OUT1;
  float* out;
};

__global__ __launch_bounds__(128) void mfn_kernel(MfnParams p) {
  // -------- LDS state & activations (~179KB) --------
  __shared__ float sActL[16 * KL_PAD];    // [x_l(300) | h_l(128) | pad]
  __shared__ float sGatesL[16 * 512];     // also reused as lastH[16][320]
  __shared__ float sActA[16 * KA_PAD];    // [x_a(74) | h_a(32) | pad]
  __shared__ float sGatesA[16 * 128];     // also g1 hidden
  __shared__ float sActV[16 * KV_PAD];    // [x_v(36) | h_v(32) | pad]
  __shared__ float sGatesV[16 * 128];     // also g2 hidden
  __shared__ float sCs[16 * TOTAL_H];     // [c_l | c_a | c_v] (persistent)
  __shared__ float sHid[16 * 256];        // shared hidden (att1 / nnp1)
  __shared__ float sAttSc[16 * TOTAL_H];  // attention logits
  __shared__ float sBoth[16 * GAMMA_IN];  // [attended(192) | mem(128)] (mem persistent)
  __shared__ float sCHat[16 * 128];
  __shared__ float sG1o[16 * 128];
  __shared__ float sG2o[16 * 128];
  __shared__ float sRedA[16 * 8];         // softmax reductions
  __shared__ float sRedB[16 * 8];

  const int tid  = threadIdx.x;
  const int lane = tid & 31;
  const int wave = tid >> 5;
  const int n0   = blockIdx.x * 16;

  // -------- init zero state + pad columns --------
  for (int i = tid; i < 16 * (KL_PAD - D_L); i += 128) {  // h_l + pad
    int r = i / (KL_PAD - D_L), d = i % (KL_PAD - D_L);
    sActL[r * KL_PAD + D_L + d] = 0.0f;
  }
  for (int i = tid; i < 16 * (KA_PAD - D_A); i += 128) {
    int r = i / (KA_PAD - D_A), d = i % (KA_PAD - D_A);
    sActA[r * KA_PAD + D_A + d] = 0.0f;
  }
  for (int i = tid; i < 16 * (KV_PAD - D_V); i += 128) {
    int r = i / (KV_PAD - D_V), d = i % (KV_PAD - D_V);
    sActV[r * KV_PAD + D_V + d] = 0.0f;
  }
  for (int i = tid; i < 16 * TOTAL_H; i += 128) sCs[i] = 0.0f;
  for (int i = tid; i < 16 * MEMD; i += 128) {
    int r = i >> 7, u = i & 127;
    sBoth[r * GAMMA_IN + TOTAL_H + u] = 0.0f;             // mem = 0
  }
  __syncthreads();

  for (int t = 0; t < T_STEPS; ++t) {
    // ---- load x_t tile (16 x 410 f32) into modality act buffers ----
    const float* xb = p.x + ((size_t)t * NBATCH + n0) * D_X;
    for (int i = tid; i < 16 * D_X; i += 128) {
      int r = i / D_X, d = i % D_X;
      float v = xb[(size_t)r * D_X + d];
      if (d < D_L)            sActL[r * KL_PAD + d] = v;
      else if (d < D_L + D_A) sActA[r * KA_PAD + (d - D_L)] = v;
      else                    sActV[r * KV_PAD + (d - D_L - D_A)] = v;
    }
    __syncthreads();

    // ---- LSTM gate GEMMs: gates = [x|h] @ [Wih|Whh]^T + bih + bhh ----
    wg_gemm<14, 32>(p.WL, sActL, KL_PAD, sGatesL, 512, p.bihL, p.bhhL, false, lane, wave);
    wg_gemm<4, 8>(p.WA, sActA, KA_PAD, sGatesA, 128, p.bihA, p.bhhA, false, lane, wave);
    wg_gemm<3, 8>(p.WV, sActV, KV_PAD, sGatesV, 128, p.bihV, p.bhhV, false, lane, wave);
    __syncthreads();

    // ---- LSTM cell updates (gate order: i, f, g, o) ----
    for (int i = tid; i < 16 * DH_L; i += 128) {
      int r = i >> 7, u = i & 127;
      float ig = sGatesL[r * 512 + u];
      float fg = sGatesL[r * 512 + 128 + u];
      float gg = sGatesL[r * 512 + 256 + u];
      float og = sGatesL[r * 512 + 384 + u];
      float c  = sCs[r * TOTAL_H + u];
      float cn = sigf(fg) * c + sigf(ig) * tanhf(gg);
      sCs[r * TOTAL_H + u] = cn;
      sActL[r * KL_PAD + D_L + u] = sigf(og) * tanhf(cn);
    }
    for (int i = tid; i < 16 * DH_AV; i += 128) {
      int r = i >> 5, u = i & 31;
      float ig = sGatesA[r * 128 + u];
      float fg = sGatesA[r * 128 + 32 + u];
      float gg = sGatesA[r * 128 + 64 + u];
      float og = sGatesA[r * 128 + 96 + u];
      float c  = sCs[r * TOTAL_H + 128 + u];
      float cn = sigf(fg) * c + sigf(ig) * tanhf(gg);
      sCs[r * TOTAL_H + 128 + u] = cn;
      sActA[r * KA_PAD + D_A + u] = sigf(og) * tanhf(cn);
    }
    for (int i = tid; i < 16 * DH_AV; i += 128) {
      int r = i >> 5, u = i & 31;
      float ig = sGatesV[r * 128 + u];
      float fg = sGatesV[r * 128 + 32 + u];
      float gg = sGatesV[r * 128 + 64 + u];
      float og = sGatesV[r * 128 + 96 + u];
      float c  = sCs[r * TOTAL_H + 160 + u];
      float cn = sigf(fg) * c + sigf(ig) * tanhf(gg);
      sCs[r * TOTAL_H + 160 + u] = cn;
      sActV[r * KV_PAD + D_V + u] = sigf(og) * tanhf(cn);
    }
    __syncthreads();

    // ---- attention MLP: 192 -> 256 (relu) -> 192 ----
    wg_gemm<6, 16>(p.ATT1, sCs, TOTAL_H, sHid, 256, p.attB1, nullptr, true, lane, wave);
    __syncthreads();
    wg_gemm<8, 12>(p.ATT2, sHid, 256, sAttSc, TOTAL_H, p.attB2, nullptr, false, lane, wave);
    __syncthreads();

    // ---- softmax over 192 feats (8 threads/row x 24 cols), attended = sm*cs ----
    {
      const int r = tid >> 3;
      const int sub = tid & 7;
      const int c0 = sub * 24;
      float lmax = -1e30f;
      for (int k = 0; k < 24; ++k) lmax = fmaxf(lmax, sAttSc[r * TOTAL_H + c0 + k]);
      sRedA[r * 8 + sub] = lmax;
      __syncthreads();
      float mx = sRedA[r * 8 + 0];
#pragma unroll
      for (int s = 1; s < 8; ++s) mx = fmaxf(mx, sRedA[r * 8 + s]);
      float lsum = 0.0f;
      for (int k = 0; k < 24; ++k) {
        float e = expf(sAttSc[r * TOTAL_H + c0 + k] - mx);
        sAttSc[r * TOTAL_H + c0 + k] = e;
        lsum += e;
      }
      sRedB[r * 8 + sub] = lsum;
      __syncthreads();
      float tot = 0.0f;
#pragma unroll
      for (int s = 0; s < 8; ++s) tot += sRedB[r * 8 + s];
      const float inv = 1.0f / tot;
      for (int k = 0; k < 24; ++k)
        sBoth[r * GAMMA_IN + c0 + k] =
            sAttSc[r * TOTAL_H + c0 + k] * inv * sCs[r * TOTAL_H + c0 + k];
    }
    __syncthreads();

    // ---- nn_p: attended(192) -> 256 (relu) -> 128 ----
    wg_gemm<6, 16>(p.NNP1, sBoth, GAMMA_IN, sHid, 256, p.nnpB1, nullptr, true, lane, wave);
    __syncthreads();
    wg_gemm<8, 8>(p.NNP2, sHid, 256, sCHat, 128, p.nnpB2, nullptr, false, lane, wave);
    // ---- gamma MLPs layer 1: both(320) -> 128 (relu) ----
    wg_gemm<10, 8>(p.G1A, sBoth, GAMMA_IN, sGatesA, 128, p.g1B1, nullptr, true, lane, wave);
    wg_gemm<10, 8>(p.G2A, sBoth, GAMMA_IN, sGatesV, 128, p.g2B1, nullptr, true, lane, wave);
    __syncthreads();
    // ---- gamma MLPs layer 2: 128 -> 128 ----
    wg_gemm<4, 8>(p.G1B, sGatesA, 128, sG1o, 128, p.g1B2, nullptr, false, lane, wave);
    wg_gemm<4, 8>(p.G2B, sGatesV, 128, sG2o, 128, p.g2B2, nullptr, false, lane, wave);
    __syncthreads();

    // ---- memory update: mem = sig(g1)*mem + sig(g2)*tanh(cHat) ----
    for (int i = tid; i < 16 * MEMD; i += 128) {
      int r = i >> 7, u = i & 127;
      float mo = sBoth[r * GAMMA_IN + TOTAL_H + u];
      float g1 = sigf(sG1o[r * 128 + u]);
      float g2 = sigf(sG2o[r * 128 + u]);
      float ch = tanhf(sCHat[r * 128 + u]);
      sBoth[r * GAMMA_IN + TOTAL_H + u] = g1 * mo + g2 * ch;
    }
    __syncthreads();
  }

  // -------- output head: last_hs = [h_l | h_a | h_v | mem] (320) --------
  float* lastH = sGatesL;  // reuse (16*320 <= 16*512)
  for (int i = tid; i < 16 * GAMMA_IN; i += 128) {
    int r = i / GAMMA_IN, d = i % GAMMA_IN;
    float v;
    if (d < 128)      v = sActL[r * KL_PAD + D_L + d];
    else if (d < 160) v = sActA[r * KA_PAD + D_A + (d - 128)];
    else if (d < 192) v = sActV[r * KV_PAD + D_V + (d - 160)];
    else              v = sBoth[r * GAMMA_IN + TOTAL_H + (d - 192)];
    lastH[r * GAMMA_IN + d] = v;
  }
  __syncthreads();
  wg_gemm<10, 8>(p.OUT1, lastH, GAMMA_IN, sG1o, 128, p.outB1, nullptr, true, lane, wave);
  __syncthreads();
  // final scalar head: y = outW2 . relu_hidden + outB2 (8 threads per row)
  {
    const int r = tid >> 3;
    const int sub = tid & 7;
    float s = 0.0f;
    for (int k = 0; k < 16; ++k)
      s += p.outW2[sub * 16 + k] * sG1o[r * 128 + sub * 16 + k];
    sRedA[r * 8 + sub] = s;
    __syncthreads();
    if (sub == 0) {
      float tot = p.outB2[0];
#pragma unroll
      for (int q = 0; q < 8; ++q) tot += sRedA[r * 8 + q];
      p.out[n0 + r] = tot;
    }
  }
}

extern "C" void kernel_launch(void* const* d_in, const int* in_sizes, int n_in,
                              void* d_out, int out_size, void* d_ws, size_t ws_size,
                              hipStream_t stream) {
  (void)in_sizes; (void)n_in; (void)out_size; (void)ws_size;
  // setup_inputs() dict (insertion) order, leaves flattened:
  const float* x     = (const float*)d_in[0];
  const float* WihL  = (const float*)d_in[1];
  const float* WhhL  = (const float*)d_in[2];
  const float* bihL  = (const float*)d_in[3];
  const float* bhhL  = (const float*)d_in[4];
  const float* WihA  = (const float*)d_in[5];
  const float* WhhA  = (const float*)d_in[6];
  const float* bihA  = (const float*)d_in[7];
  const float* bhhA  = (const float*)d_in[8];
  const float* WihV  = (const float*)d_in[9];
  const float* WhhV  = (const float*)d_in[10];
  const float* bihV  = (const float*)d_in[11];
  const float* bhhV  = (const float*)d_in[12];
  const float* attW1 = (const float*)d_in[13];
  const float* attB1 = (const float*)d_in[14];
  const float* attW2 = (const float*)d_in[15];
  const float* attB2 = (const float*)d_in[16];
  const float* nnpW1 = (const float*)d_in[17];
  const float* nnpB1 = (const float*)d_in[18];
  const float* nnpW2 = (const float*)d_in[19];
  const float* nnpB2 = (const float*)d_in[20];
  const float* g1W1  = (const float*)d_in[21];
  const float* g1B1  = (const float*)d_in[22];
  const float* g1W2  = (const float*)d_in[23];
  const float* g1B2  = (const float*)d_in[24];
  const float* g2W1  = (const float*)d_in[25];
  const float* g2B1  = (const float*)d_in[26];
  const float* g2W2  = (const float*)d_in[27];
  const float* g2B2  = (const float*)d_in[28];
  const float* outW1 = (const float*)d_in[29];
  const float* outB1 = (const float*)d_in[30];
  const float* outW2 = (const float*)d_in[31];
  const float* outB2 = (const float*)d_in[32];

  // workspace: packed f16 weight fragments (256 dwords each). ~1.2 MB total.
  uint32_t* ws = (uint32_t*)d_ws;
  size_t off = 0;
  auto alloc = [&](int nt, int kt) { uint32_t* q = ws + off; off += (size_t)nt * kt * 256; return q; };
  uint32_t* WL   = alloc(32, 14);
  uint32_t* WA   = alloc(8, 4);
  uint32_t* WV   = alloc(8, 3);
  uint32_t* ATT1 = alloc(16, 6);
  uint32_t* ATT2 = alloc(12, 8);
  uint32_t* NNP1 = alloc(16, 6);
  uint32_t* NNP2 = alloc(8, 8);
  uint32_t* G1A  = alloc(8, 10);
  uint32_t* G1B  = alloc(8, 4);
  uint32_t* G2A  = alloc(8, 10);
  uint32_t* G2B  = alloc(8, 4);
  uint32_t* OUT1 = alloc(8, 10);

  // pack (deterministic; rerun every call since ws is not preserved)
  pack_weights<<<32 * 14, 32, 0, stream>>>(WihL, 300, WhhL, 128, 512, 14, WL);
  pack_weights<<<8 * 4, 32, 0, stream>>>(WihA, 74, WhhA, 32, 128, 4, WA);
  pack_weights<<<8 * 3, 32, 0, stream>>>(WihV, 36, WhhV, 32, 128, 3, WV);
  pack_weights<<<16 * 6, 32, 0, stream>>>(attW1, 192, nullptr, 0, 256, 6, ATT1);
  pack_weights<<<12 * 8, 32, 0, stream>>>(attW2, 256, nullptr, 0, 192, 8, ATT2);
  pack_weights<<<16 * 6, 32, 0, stream>>>(nnpW1, 192, nullptr, 0, 256, 6, NNP1);
  pack_weights<<<8 * 8, 32, 0, stream>>>(nnpW2, 256, nullptr, 0, 128, 8, NNP2);
  pack_weights<<<8 * 10, 32, 0, stream>>>(g1W1, 320, nullptr, 0, 128, 10, G1A);
  pack_weights<<<8 * 4, 32, 0, stream>>>(g1W2, 128, nullptr, 0, 128, 4, G1B);
  pack_weights<<<8 * 10, 32, 0, stream>>>(g2W1, 320, nullptr, 0, 128, 10, G2A);
  pack_weights<<<8 * 4, 32, 0, stream>>>(g2W2, 128, nullptr, 0, 128, 4, G2B);
  pack_weights<<<8 * 10, 32, 0, stream>>>(outW1, 320, nullptr, 0, 128, 10, OUT1);

  MfnParams p;
  p.x = x;
  p.bihL = bihL; p.bhhL = bhhL; p.bihA = bihA; p.bhhA = bhhA; p.bihV = bihV; p.bhhV = bhhV;
  p.attB1 = attB1; p.attB2 = attB2; p.nnpB1 = nnpB1; p.nnpB2 = nnpB2;
  p.g1B1 = g1B1; p.g1B2 = g1B2; p.g2B1 = g2B1; p.g2B2 = g2B2;
  p.outB1 = outB1; p.outW2 = outW2; p.outB2 = outB2;
  p.WL = WL; p.WA = WA; p.WV = WV; p.ATT1 = ATT1; p.ATT2 = ATT2;
  p.NNP1 = NNP1; p.NNP2 = NNP2; p.G1A = G1A; p.G1B = G1B; p.G2A = G2A; p.G2B = G2B;
  p.OUT1 = OUT1;
  p.out = (float*)d_out;

  mfn_kernel<<<NBATCH / 16, 128, 0, stream>>>(p);
}
